// Critic_PreAttentionGRU_3616362463946
// MI455X (gfx1250) — matrix operand order repaired and checked by reading, not compiled, not run on hardware
//
#include <hip/hip_runtime.h>

// ---------------------------------------------------------------------------
// CDNA5 (gfx1250) implementation. wave32, WMMA f32_16x16x32_f16 everywhere,
// async global->LDS weight staging, native rcp/tanh transcendentals.
// ---------------------------------------------------------------------------

typedef __attribute__((ext_vector_type(16))) _Float16 v16h;
typedef __attribute__((ext_vector_type(8)))  _Float16 v8h;
typedef __attribute__((ext_vector_type(8)))  float    v8f;
typedef int v4i __attribute__((vector_size(16)));

#define HH 32
#define BB 2048
#define TT 128
#define NN (BB*TT)   // 262144 tokens

#define TILES_BYTES 172032            // 168 tiles * 512 f16 * 2B
#define BIASN 912                     // 256+256+256+96+32+1 (+pad to 16B)
#define BLOB_BYTES (TILES_BYTES + BIASN*4)   // 175680, 16B multiple
#define SCR_BYTES (8*4096*2)          // per-block wave scratch (8 waves x 8KB)

// A-operand (16x32 f16): lane holds row m = lane&15.
// lanes 0-15: K0..7 then K16..23 ; lanes 16-31: K8..15 then K24..31
__device__ __forceinline__ int kmapA(int lane, int j) {
    return (j & 7) + ((j & 8) ? 16 : 0) + ((lane & 16) ? 8 : 0);
}
// B-operand (32x16 f16): lane holds col n = lane&15; lanes 0-15 K0..7 then K16..23 ; lanes 16-31 K8..15 then K24..31
__device__ __forceinline__ int kmapB(int lane, int j) {
    return (j & 7) + ((j & 8) ? 16 : 0) + ((lane & 16) ? 8 : 0);
}

__device__ __forceinline__ v8f wmma16(v16h a, v16h b, v8f c) {
    return __builtin_amdgcn_wmma_f32_16x16x32_f16(false, a, false, b, (short)0, c, false, false);
}
__device__ __forceinline__ v8f splat8(float x) {
    v8f c;
    #pragma unroll
    for (int r = 0; r < 8; ++r) c[r] = x;
    return c;
}

// ---- fast transcendentals: v_rcp_f32 / v_tanh_f32 (CDNA5 TRANS ops) --------
__device__ __forceinline__ float fast_rcp(float x) {
#if defined(__has_builtin) && __has_builtin(__builtin_amdgcn_rcpf)
    return __builtin_amdgcn_rcpf(x);
#else
    return 1.0f / x;
#endif
}
__device__ __forceinline__ float sigm(float x) {
    return fast_rcp(1.0f + __expf(-x));
}
__device__ __forceinline__ float fast_tanh(float x) {
#if defined(__has_builtin) && __has_builtin(__builtin_amdgcn_tanhf)
    return __builtin_amdgcn_tanhf(x);
#else
    return 1.0f - 2.0f * fast_rcp(__expf(2.0f * x) + 1.0f);
#endif
}
__device__ __forceinline__ float leaky(float x) { return x > 0.0f ? x : 0.01f * x; }
__device__ __forceinline__ bool  iscl(float a, float b) {
    return fabsf(a - b) <= (1e-8f + 1e-5f * fabsf(b));
}

// A-pack from an f16 row in LDS (two 16B vector loads, no converts)
__device__ __forceinline__ v16h packA_lds(const _Float16* rowbase, int lane) {
    const int kb = (lane & 16) ? 8 : 0;
    v8h lo = *(const v8h*)(rowbase + kb);
    v8h hi = *(const v8h*)(rowbase + kb + 16);
    v16h r;
    #pragma unroll
    for (int j = 0; j < 8; ++j) { r[j] = lo[j]; r[j + 8] = hi[j]; }
    return r;
}
// A-pack from an f32 row in global memory (contiguous chunks -> b128 loads)
__device__ __forceinline__ v16h packA_g32(const float* rowbase, int lane) {
    const int kb = (lane & 16) ? 8 : 0;
    v16h r;
    #pragma unroll
    for (int j = 0; j < 8; ++j) {
        r[j]     = (_Float16)rowbase[kb + j];
        r[j + 8] = (_Float16)rowbase[kb + 16 + j];
    }
    return r;
}

// ---- CDNA5 async global->LDS copy (ASYNCcnt path), with safe fallback ------
#if defined(__has_builtin)
#if __has_builtin(__builtin_amdgcn_global_load_async_to_lds_b128) && \
    __has_builtin(__builtin_amdgcn_s_wait_asynccnt)
#define USE_ASYNC_LDS 1
#endif
#endif

__device__ __forceinline__ void cp16_lds(void* lds, const void* g) {
#ifdef USE_ASYNC_LDS
    __builtin_amdgcn_global_load_async_to_lds_b128(
        (__attribute__((address_space(1))) v4i*)(void*)g,
        (__attribute__((address_space(3))) v4i*)lds, 0, 0);
#else
    *(int4*)lds = *(const int4*)g;
#endif
}
__device__ __forceinline__ void cp_lds_wait() {
#ifdef USE_ASYNC_LDS
    __builtin_amdgcn_s_wait_asynccnt(0);
#endif
}

// ---------------------------------------------------------------------------
// K1: fold encoder into GRU input matmul:  W' = Wih @ encW  (96x7 / 96x4),
//     b' = encb @ Wih.T + bih.
// ---------------------------------------------------------------------------
__global__ void setup_kernel(const float* __restrict__ encAW, const float* __restrict__ encAb,
                             const float* __restrict__ encMW, const float* __restrict__ encMb,
                             const float* __restrict__ aWih,  const float* __restrict__ abih,
                             const float* __restrict__ mWih,  const float* __restrict__ mbih,
                             float* __restrict__ WpA, float* __restrict__ bpA,
                             float* __restrict__ WpM, float* __restrict__ bpM) {
    int g = threadIdx.x;
    if (g >= 96) return;
    float bacc = abih[g];
    for (int j = 0; j < 32; ++j) bacc += aWih[g*32 + j] * encAb[j];
    bpA[g] = bacc;
    for (int i = 0; i < 7; ++i) {
        float a = 0.f;
        for (int j = 0; j < 32; ++j) a += aWih[g*32 + j] * encAW[j*7 + i];
        WpA[g*7 + i] = a;
    }
    bacc = mbih[g];
    for (int j = 0; j < 32; ++j) bacc += mWih[g*32 + j] * encMb[j];
    bpM[g] = bacc;
    for (int i = 0; i < 4; ++i) {
        float a = 0.f;
        for (int j = 0; j < 32; ++j) a += mWih[g*32 + j] * encMW[j*4 + i];
        WpM[g*4 + i] = a;
    }
}

// ---------------------------------------------------------------------------
// K1b: pre-pack attention/MLP weights + biases into an LDS-image blob (ws).
//      Layout must match attn_mlp_kernel's LDS exactly.
// ---------------------------------------------------------------------------
__global__ __launch_bounds__(256) void pack_kernel(
    const float* __restrict__ attn_in_w, const float* __restrict__ attn_in_b,
    const float* __restrict__ attn_out_w, const float* __restrict__ attn_out_b,
    const float* __restrict__ mlp_W0, const float* __restrict__ mlp_b0,
    const float* __restrict__ mlp_W1, const float* __restrict__ mlp_b1,
    const float* __restrict__ out_W, const float* __restrict__ out_b,
    char* __restrict__ blob) {
    int e = blockIdx.x * 256 + threadIdx.x;
    _Float16* tiles = (_Float16*)blob;
    float* bias = (float*)(blob + TILES_BYTES);
    if (e < 168*512) {
        int tt = e >> 9, q = e & 511;
        int plane = q >> 4, j = q & 15;
        int nl = plane & 15;
        int k  = kmapB(plane, j);
        float w;
        if (tt < 8) {                     // 0,1:Wq 2,3:Wk 4,5:Wv 6,7:attn_out
            int mat = tt >> 1, sub = tt & 1;
            int n = sub*16 + nl;
            w = (mat < 3) ? attn_in_w[(mat*32 + n)*32 + k]
                          : attn_out_w[n*32 + k];
        } else if (tt < 40) {             // W0: [nt(16)][kt(2)]
            int idx = tt - 8, nt = idx >> 1, kt = idx & 1;
            w = mlp_W0[(nt*16 + nl)*64 + kt*32 + k];
        } else {                          // W1: [nt(16)][kt(8)]
            int idx = tt - 40, nt = idx >> 3, kt = idx & 7;
            w = mlp_W1[(nt*16 + nl)*256 + kt*32 + k];
        }
        tiles[e] = (_Float16)w;
    } else if (e < 168*512 + BIASN) {
        int b = e - 168*512;
        float v = 0.f;
        if (b < 256)      v = mlp_b0[b];
        else if (b < 512) v = mlp_b1[b - 256];
        else if (b < 768) v = out_W[b - 512];
        else if (b < 864) v = attn_in_b[b - 768];
        else if (b < 896) v = attn_out_b[b - 864];
        else if (b == 896) v = out_b[0];
        bias[b] = v;
    }
}

// ---------------------------------------------------------------------------
// K2: precompute x-gates for both GRUs (f16), off the sequential critical path.
// ---------------------------------------------------------------------------
template<int K>
__device__ __forceinline__ void enc_token(const float* o, const float* W,
                                          const float* b, _Float16* out) {
    float ov[K];
    #pragma unroll
    for (int k = 0; k < K; ++k) ov[k] = o[k];
    #pragma unroll
    for (int gb = 0; gb < 6; ++gb) {      // 6 chunks of 16 -> 32B vector stores
        v16h chunk;
        #pragma unroll
        for (int gi = 0; gi < 16; ++gi) {
            int g = gb*16 + gi;
            float acc = b[g];
            #pragma unroll
            for (int k = 0; k < K; ++k) acc += W[g*K + k] * ov[k];
            chunk[gi] = (_Float16)acc;
        }
        *(v16h*)(out + gb*16) = chunk;
    }
}

__global__ __launch_bounds__(256) void encode_kernel(
    const float* __restrict__ obs,
    const float* __restrict__ WpA, const float* __restrict__ bpA,
    const float* __restrict__ WpM, const float* __restrict__ bpM,
    _Float16* __restrict__ x_air, _Float16* __restrict__ x_m) {
    __shared__ float sW[1248];
    const int tid = threadIdx.x;
    for (int i = tid; i < 1248; i += 256) {
        float v;
        if (i < 672)       v = WpA[i];
        else if (i < 768)  v = bpA[i - 672];
        else if (i < 1152) v = WpM[i - 768];
        else               v = bpM[i - 1152];
        sW[i] = v;
    }
    __syncthreads();
    const float* sWpA = sW;
    const float* sbpA = sW + 672;
    const float* sWpM = sW + 768;
    const float* sbpM = sW + 1152;

    size_t u = (size_t)blockIdx.x * 256 + tid;
    if (u < (size_t)NN) {
        enc_token<7>(obs + u*15 + 8, sWpA, sbpA, x_air + u*96);
    } else {
        size_t v = u - (size_t)NN;                 // m-token index in [0, 2N)
        size_t obsTok = (v < (size_t)NN) ? v : v - (size_t)NN;
        int coff = (v < (size_t)NN) ? 0 : 4;
        enc_token<4>(obs + obsTok*15 + coff, sWpM, sbpM, x_m + v*96);
    }
}

// ---------------------------------------------------------------------------
// K3: the two GRUs. One wave per block, 16 batch rows per wave, full T loop.
//     h staged in LDS as f16 (WMMA A is f16 anyway), x-gates loaded as v16h.
// ---------------------------------------------------------------------------
__global__ __launch_bounds__(32) void gru_kernel(
    const float* __restrict__ rnn_state,
    const float* __restrict__ air_Whh, const float* __restrict__ air_bhh,
    const float* __restrict__ m_Whh,   const float* __restrict__ m_bhh,
    const _Float16* __restrict__ x_air, const _Float16* __restrict__ x_m,
    float* __restrict__ feat_air, float* __restrict__ feat_m,
    float* __restrict__ next_h) {
    __shared__ float gh[16*96];
    __shared__ _Float16 hbuf16[16*32];

    const int lane = threadIdx.x;
    const int ml = lane & 15;
    const int c0 = (lane & 16) ? 16 : 0;   // this lane's h-column window
    const int mb = (lane & 16) ? 8 : 0;    // D-layout row base

    const bool isAir = blockIdx.x < (BB/16);
    const int rbase  = isAir ? blockIdx.x*16 : (blockIdx.x - BB/16)*16;
    const float* Whh = isAir ? air_Whh : m_Whh;
    const float* bhh = isAir ? air_bhh : m_bhh;
    const _Float16* xp = isAir ? x_air : x_m;
    float* feat = isAir ? feat_air : feat_m;
    const int row = rbase + ml;

    // Recurrent weights as 6 WMMA-B tiles, resident in VGPRs for all 128 steps
    v16h Bt[6];
    float bt[6];
    #pragma unroll
    for (int nt = 0; nt < 6; ++nt) {
        int n = nt*16 + ml;
        #pragma unroll
        for (int j = 0; j < 16; ++j)
            Bt[nt][j] = (_Float16)Whh[n*32 + kmapB(lane, j)];
        bt[nt] = bhh[n];
    }

    // init h (reference's interleaved reshape for the missile state)
    float hreg[16];
    {
        v16h hv;
        #pragma unroll
        for (int c = 0; c < 16; ++c) {
            int col = c0 + c;
            float h0 = isAir ? rnn_state[(size_t)row*96 + col]
                             : rnn_state[(size_t)(row >> 1)*96 + 32 + (row & 1)*32 + col];
            hreg[c] = h0;
            hv[c] = (_Float16)h0;
        }
        *(v16h*)(hbuf16 + ml*32 + c0) = hv;
    }
    __syncthreads();

    for (int t = 0; t < TT; ++t) {
        // x-gates for this step: three contiguous 32B chunks per lane
        const _Float16* xrow = xp + ((size_t)row*TT + t)*96;
        if (t + 1 < TT) __builtin_prefetch(xrow + 96, 0, 0);
        v16h xr16 = *(const v16h*)(xrow + c0);
        v16h xz16 = *(const v16h*)(xrow + 32 + c0);
        v16h xn16 = *(const v16h*)(xrow + 64 + c0);

        // A-pack h from f16 LDS buffer
        v16h A = packA_lds(hbuf16 + ml*32, lane);
        // gh = h @ Whh.T + bhh : 6 x wmma
        #pragma unroll
        for (int nt = 0; nt < 6; ++nt) {
            v8f d = wmma16(A, Bt[nt], splat8(bt[nt]));
            #pragma unroll
            for (int r = 0; r < 8; ++r)
                gh[(mb + r)*96 + nt*16 + ml] = d[r];
        }
        __syncthreads();

        #pragma unroll
        for (int c = 0; c < 16; ++c) {
            int col = c0 + c;
            float hr = gh[ml*96 + col];
            float hz = gh[ml*96 + 32 + col];
            float hn = gh[ml*96 + 64 + col];
            float r  = sigm((float)xr16[c] + hr);
            float z  = sigm((float)xz16[c] + hz);
            float nn = fast_tanh((float)xn16[c] + r*hn);
            hreg[c] = (1.f - z)*nn + z*hreg[c];
        }
        __syncthreads();
        {
            v16h hv;
            #pragma unroll
            for (int c = 0; c < 16; ++c) hv[c] = (_Float16)hreg[c];
            *(v16h*)(hbuf16 + ml*32 + c0) = hv;
        }

        float* fo = feat + ((size_t)row*TT + t)*32 + c0;
        #pragma unroll
        for (int c4 = 0; c4 < 4; ++c4)
            ((float4*)fo)[c4] = make_float4(hreg[4*c4], hreg[4*c4+1], hreg[4*c4+2], hreg[4*c4+3]);
        __syncthreads();
    }

    // final hidden state -> d_out[N ...]
    #pragma unroll
    for (int c = 0; c < 16; ++c) {
        int col = c0 + c;
        if (isAir) next_h[(size_t)row*96 + col] = hreg[c];
        else       next_h[(size_t)(row >> 1)*96 + 32 + (row & 1)*32 + col] = hreg[c];
    }
}

// ---------------------------------------------------------------------------
// K4: attention + MLP. 8 waves/block, 16 tokens/wave, 128 tokens/block.
//     Weights arrive as a pre-packed LDS image: one contiguous async
//     global->LDS copy per block (ASYNCcnt path on CDNA5).
// ---------------------------------------------------------------------------
__global__ __launch_bounds__(256) void attn_mlp_kernel(
    const float* __restrict__ obs,
    const float* __restrict__ feat_air, const float* __restrict__ feat_m,
    const char* __restrict__ blob,
    float* __restrict__ outv) {
    extern __shared__ char smem_raw[];
    _Float16* tiles = (_Float16*)smem_raw;
    float* bias = (float*)(smem_raw + TILES_BYTES);
    float* sb0 = bias;             // 256
    float* sb1 = bias + 256;       // 256
    float* sow = bias + 512;       // 256
    float* sab = bias + 768;       // 96
    float* sob = bias + 864;       // 32  (bias[896] = out_b)
    _Float16* scrAll = (_Float16*)(smem_raw + BLOB_BYTES);

    const int tid = threadIdx.x;

    // ---- stage the whole weight image into LDS (16B per lane per issue)
    for (int i = tid; i < BLOB_BYTES / 16; i += 256)
        cp16_lds(smem_raw + (size_t)i*16, blob + (size_t)i*16);
    cp_lds_wait();
    __syncthreads();

    const int lane = tid & 31;
    const int wv   = tid >> 5;
    const int tok0 = blockIdx.x * 128 + wv * 16;
    _Float16* scr  = scrAll + wv * 4096;
    const int ml = lane & 15;
    const int mb = (lane & 16) ? 8 : 0;

    // ---- A-packs of the three feature streams (K=32)
    v16h Aair = packA_g32(feat_air + (size_t)(tok0 + ml)*32, lane);
    v16h Am1  = packA_g32(feat_m   + (size_t)(tok0 + ml)*32, lane);
    v16h Am2  = packA_g32(feat_m   + (size_t)(NN + tok0 + ml)*32, lane);

    // ---- q/k/v (10 WMMAs) -> scratch, token stride 200 f16
    #pragma unroll
    for (int s = 0; s < 2; ++s) {
        v16h bq = *(const v16h*)(tiles + (0 + s)*512 + lane*16);
        v16h bk = *(const v16h*)(tiles + (2 + s)*512 + lane*16);
        v16h bv = *(const v16h*)(tiles + (4 + s)*512 + lane*16);
        v8f dq  = wmma16(Aair, bq, splat8(sab[s*16 + ml]));
        v8f dk1 = wmma16(Am1,  bk, splat8(sab[32 + s*16 + ml]));
        v8f dk2 = wmma16(Am2,  bk, splat8(sab[32 + s*16 + ml]));
        v8f dv1 = wmma16(Am1,  bv, splat8(sab[64 + s*16 + ml]));
        v8f dv2 = wmma16(Am2,  bv, splat8(sab[64 + s*16 + ml]));
        #pragma unroll
        for (int r = 0; r < 8; ++r) {
            int o = (mb + r)*200 + s*16 + ml;
            scr[o +   0] = (_Float16)dq[r];
            scr[o +  32] = (_Float16)dk1[r];
            scr[o +  64] = (_Float16)dk2[r];
            scr[o +  96] = (_Float16)dv1[r];
            scr[o + 128] = (_Float16)dv2[r];
        }
    }
    __syncthreads();

    // ---- scores / softmax(2) / ctx : lane -> (token ml, head hh)
    {
        int hh = (lane >> 4) & 1;
        const _Float16* base = scr + ml*200;
        float s1 = 0.f, s2 = 0.f;
        #pragma unroll
        for (int d = 0; d < 16; ++d) {
            float qd = (float)base[hh*16 + d];
            s1 += qd * (float)base[32 + hh*16 + d];
            s2 += qd * (float)base[64 + hh*16 + d];
        }
        s1 *= 0.25f; s2 *= 0.25f;
        const float* o = obs + (size_t)(tok0 + ml) * 15;
        bool mk1 = iscl(o[0],1.f) && iscl(o[1],0.f) && iscl(o[2],1.f) && iscl(o[3],0.f);
        bool mk2 = iscl(o[4],1.f) && iscl(o[5],0.f) && iscl(o[6],1.f) && iscl(o[7],0.f);
        if (mk1) s1 += -1.0e9f;
        if (mk2) s2 += -1.0e9f;
        float mx = fmaxf(s1, s2);
        float e1 = __expf(s1 - mx), e2 = __expf(s2 - mx);
        float inv = fast_rcp(e1 + e2);
        float w1 = e1 * inv, w2 = e2 * inv;
        _Float16* cdst = scr + ml*200 + 160 + hh*16;
        #pragma unroll
        for (int d = 0; d < 16; ++d) {
            float cv = w1 * (float)base[96 + hh*16 + d] + w2 * (float)base[128 + hh*16 + d];
            cdst[d] = (_Float16)cv;
        }
        if (hh == 0) scr[ml*200 + 192] = (_Float16)((mk1 && mk2) ? 0.f : 1.f);
    }
    __syncthreads();

    // ---- attn = ctx @ attn_out.T + b, zero when both masked
    v16h Actx = packA_lds(scr + ml*200 + 160, lane);
    #pragma unroll
    for (int s = 0; s < 2; ++s) {
        v16h bo = *(const v16h*)(tiles + (6 + s)*512 + lane*16);
        v8f d = wmma16(Actx, bo, splat8(sob[s*16 + ml]));
        #pragma unroll
        for (int r = 0; r < 8; ++r) {
            float fl = (float)scr[(mb + r)*200 + 192];
            scr[(mb + r)*200 + s*16 + ml] = (_Float16)(d[r] * fl);
        }
    }
    __syncthreads();
    v16h Aattn = packA_lds(scr + ml*200, lane);
    __syncthreads();

    // ---- h1 = leaky([air|attn] @ W0.T + b0)  (32 WMMAs)
    #pragma unroll
    for (int nt = 0; nt < 16; ++nt) {
        v8f d = splat8(sb0[nt*16 + ml]);
        d = wmma16(Aair,  *(const v16h*)(tiles + (8 + nt*2 + 0)*512 + lane*16), d);
        d = wmma16(Aattn, *(const v16h*)(tiles + (8 + nt*2 + 1)*512 + lane*16), d);
        #pragma unroll
        for (int r = 0; r < 8; ++r)
            scr[(mb + r)*256 + nt*16 + ml] = (_Float16)leaky(d[r]);
    }
    __syncthreads();
    v16h Ah[8];
    #pragma unroll
    for (int kt = 0; kt < 8; ++kt)
        Ah[kt] = packA_lds(scr + ml*256 + kt*32, lane);
    __syncthreads();

    // ---- h2 = leaky(h1 @ W1.T + b1)  (128 WMMAs)
    #pragma unroll
    for (int nt = 0; nt < 16; ++nt) {
        v8f d = splat8(sb1[nt*16 + ml]);
        #pragma unroll
        for (int kt = 0; kt < 8; ++kt)
            d = wmma16(Ah[kt], *(const v16h*)(tiles + (40 + nt*8 + kt)*512 + lane*16), d);
        #pragma unroll
        for (int r = 0; r < 8; ++r)
            scr[(mb + r)*256 + nt*16 + ml] = (_Float16)leaky(d[r]);
    }
    __syncthreads();

    // ---- val = h2 @ out_W.T + out_b : split 256-dot across two half-waves
    {
        int half = (lane >> 4) & 1;
        float acc = 0.f;
        const v8h* hp = (const v8h*)(scr + ml*256 + half*128);
        #pragma unroll
        for (int c8 = 0; c8 < 16; ++c8) {
            v8h hv = hp[c8];
            #pragma unroll
            for (int j = 0; j < 8; ++j)
                acc += (float)hv[j] * sow[half*128 + c8*8 + j];
        }
        acc += __shfl_xor(acc, 16);
        if (lane < 16) outv[tok0 + ml] = acc + bias[896];
    }
}

// ---------------------------------------------------------------------------
// host launcher
// ---------------------------------------------------------------------------
extern "C" void kernel_launch(void* const* d_in, const int* in_sizes, int n_in,
                              void* d_out, int out_size, void* d_ws, size_t ws_size,
                              hipStream_t stream) {
    const float* obs        = (const float*)d_in[0];
    const float* rnn_state  = (const float*)d_in[1];
    const float* enc_air_W  = (const float*)d_in[2];
    const float* enc_air_b  = (const float*)d_in[3];
    const float* enc_m_W    = (const float*)d_in[4];
    const float* enc_m_b    = (const float*)d_in[5];
    const float* air_Wih    = (const float*)d_in[6];
    const float* air_Whh    = (const float*)d_in[7];
    const float* air_bih    = (const float*)d_in[8];
    const float* air_bhh    = (const float*)d_in[9];
    const float* m_Wih      = (const float*)d_in[10];
    const float* m_Whh      = (const float*)d_in[11];
    const float* m_bih      = (const float*)d_in[12];
    const float* m_bhh      = (const float*)d_in[13];
    const float* attn_in_w  = (const float*)d_in[14];
    const float* attn_in_b  = (const float*)d_in[15];
    const float* attn_out_w = (const float*)d_in[16];
    const float* attn_out_b = (const float*)d_in[17];
    const float* mlp_W0     = (const float*)d_in[18];
    const float* mlp_b0     = (const float*)d_in[19];
    const float* mlp_W1     = (const float*)d_in[20];
    const float* mlp_b1     = (const float*)d_in[21];
    const float* out_W      = (const float*)d_in[22];
    const float* out_b      = (const float*)d_in[23];

    float* wsf = (float*)d_ws;
    float* WpA = wsf;                    // 96*7
    float* bpA = wsf + 672;              // 96
    float* WpM = wsf + 768;              // 96*4
    float* bpM = wsf + 1152;             // 96
    float* feat_air = wsf + 1280;                    // N*32 f32
    float* feat_m   = feat_air + (size_t)NN * 32;    // 2N*32 f32
    _Float16* x_air = (_Float16*)(feat_m + (size_t)2 * NN * 32);  // N*96 f16
    _Float16* x_m   = x_air + (size_t)NN * 96;                    // 2N*96 f16
    char* blob      = (char*)(x_m + (size_t)2 * NN * 96);         // LDS image

    float* outv   = (float*)d_out;       // N values
    float* next_h = outv + NN;           // B*96

    setup_kernel<<<1, 128, 0, stream>>>(enc_air_W, enc_air_b, enc_m_W, enc_m_b,
                                        air_Wih, air_bih, m_Wih, m_bih,
                                        WpA, bpA, WpM, bpM);
    pack_kernel<<<(168*512 + BIASN + 255) / 256, 256, 0, stream>>>(
        attn_in_w, attn_in_b, attn_out_w, attn_out_b,
        mlp_W0, mlp_b0, mlp_W1, mlp_b1, out_W, out_b, blob);
    encode_kernel<<<3 * NN / 256, 256, 0, stream>>>(obs, WpA, bpA, WpM, bpM, x_air, x_m);
    gru_kernel<<<BB/16 + 2*BB/16, 32, 0, stream>>>(rnn_state,
                                                   air_Whh, air_bhh, m_Whh, m_bhh,
                                                   x_air, x_m, feat_air, feat_m, next_h);
    size_t shmem = (size_t)BLOB_BYTES + SCR_BYTES;   // ~241 KB (<320KB/WGP)
    attn_mlp_kernel<<<NN / 128, 256, shmem, stream>>>(obs, feat_air, feat_m,
                                                      blob, outv);
}
